// AttentionNoPairBias_37546604101754
// MI455X (gfx1250) — compile-verified, matrix operand order checked by reading, not run on hardware
//
#include <hip/hip_runtime.h>

// Problem constants (B=1)
#define SQ 2048   // query length S
#define NK 2048   // key length N
#define CH 1024   // channels C
#define NH 16     // heads
#define DHD 64    // head dim

typedef __attribute__((ext_vector_type(16))) __bf16 v16bf;
typedef __attribute__((ext_vector_type(8)))  float  v8f;
typedef __attribute__((ext_vector_type(4)))  unsigned int v4u;
typedef __attribute__((ext_vector_type(8)))  int v8i;
typedef __attribute__((ext_vector_type(4)))  int v4i;

struct Frag32B { uint4 lo, hi; };

__device__ __forceinline__ v16bf load_frag(const unsigned short* p) {
  Frag32B t;
  t.lo = *reinterpret_cast<const uint4*>(p);
  t.hi = *reinterpret_cast<const uint4*>(p + 8);
  return __builtin_bit_cast(v16bf, t);
}

__device__ __forceinline__ unsigned short f2bf(float f) {
  unsigned int u = __builtin_bit_cast(unsigned int, f);
  u += 0x7FFFu + ((u >> 16) & 1u);
  return (unsigned short)(u >> 16);
}

__device__ __forceinline__ v8f wmma_bf16(v16bf a, v16bf b, v8f c) {
  return __builtin_amdgcn_wmma_f32_16x16x32_bf16(false, a, false, b, (short)0, c,
                                                 false, false);
}

// 16-lane butterfly reductions via DPP16 ROW_XMASK: pure VALU, lanes 0-15 and
// 16-31 reduce independently, matching the WMMA C/D half-row layout.
template <int CTRL>
__device__ __forceinline__ float dpp_xor(float x) {
  int v = __builtin_bit_cast(int, x);
  int r = __builtin_amdgcn_update_dpp(v, v, CTRL, 0xF, 0xF, false);
  return __builtin_bit_cast(float, r);
}
__device__ __forceinline__ float rowmax16(float x) {
  x = fmaxf(x, dpp_xor<0x161>(x));
  x = fmaxf(x, dpp_xor<0x162>(x));
  x = fmaxf(x, dpp_xor<0x164>(x));
  x = fmaxf(x, dpp_xor<0x168>(x));
  return x;
}
__device__ __forceinline__ float rowsum16(float x) {
  x += dpp_xor<0x161>(x);
  x += dpp_xor<0x162>(x);
  x += dpp_xor<0x164>(x);
  x += dpp_xor<0x168>(x);
  return x;
}

__device__ __forceinline__ float fexp2(float x) { return __builtin_amdgcn_exp2f(x); }

// Async copy of 64 contiguous bytes global -> LDS (4 x b128), ASYNCcnt-tracked.
__device__ __forceinline__ void async_copy64(unsigned lds_off,
                                             const unsigned short* gptr) {
  unsigned long long ga = (unsigned long long)gptr;
  asm volatile(
      "global_load_async_to_lds_b128 %0, %1, off\n\t"
      "global_load_async_to_lds_b128 %0, %1, off offset:16\n\t"
      "global_load_async_to_lds_b128 %0, %1, off offset:32\n\t"
      "global_load_async_to_lds_b128 %0, %1, off offset:48"
      :: "v"(lds_off), "v"(ga) : "memory");
}
__device__ __forceinline__ void wait_async0() {
  asm volatile("s_wait_asynccnt 0x0" ::: "memory");
}

// TDM: 2D bf16 tile (tile_d0 x tile_d1) global -> LDS, with 16B row padding
// after every 64 elements (pad_interval=32 dwords -> code 4, pad_amount=4
// dwords -> code 3) reproducing the [rows][72] LDS layout. TENSORcnt-tracked.
__device__ __forceinline__ void tdm_load_2d(unsigned lds_addr, const void* gaddr,
                                            unsigned tensor_d0, unsigned tensor_d1,
                                            unsigned tile_d0, unsigned tile_d1,
                                            unsigned stride0) {
  unsigned long long ga = (unsigned long long)gaddr;
  v4u g0;
  g0[0] = 1u;                                             // count=1, user D#
  g0[1] = lds_addr;                                       // lds_addr (bytes)
  g0[2] = (unsigned)(ga & 0xFFFFFFFFu);                   // global_addr[31:0]
  g0[3] = (unsigned)((ga >> 32) & 0x1FFFFFFu) | (2u << 30);  // addr[56:32], type=2
  v8i g1;
  g1[0] = (int)((1u << 16) | (1u << 20) | (4u << 22) | (3u << 25));
  g1[1] = (int)((tensor_d0 & 0xFFFFu) << 16);
  g1[2] = (int)(((tensor_d0 >> 16) & 0xFFFFu) | ((tensor_d1 & 0xFFFFu) << 16));
  g1[3] = (int)(((tensor_d1 >> 16) & 0xFFFFu) | ((tile_d0 & 0xFFFFu) << 16));
  g1[4] = (int)(tile_d1 & 0xFFFFu);                       // tile_dim1, tile_dim2=0
  g1[5] = (int)stride0;                                   // dim0 stride (elems)
  g1[6] = 0;
  g1[7] = 0;
  v4i z4 = {0, 0, 0, 0};
  v8i z8 = {0, 0, 0, 0, 0, 0, 0, 0};
  __builtin_amdgcn_tensor_load_to_lds(g0, g1, z4, z4, z8, 0);
}
__device__ __forceinline__ void wait_tensor0() {
  __builtin_amdgcn_s_wait_tensorcnt((short)0);
}

// ---------------------------------------------------------------------------
// fp32 -> bf16 conversion (RNE)
// ---------------------------------------------------------------------------
__global__ void __launch_bounds__(256)
cvt_f32_bf16(const float* __restrict__ in, unsigned short* __restrict__ out, int n) {
  int i = (blockIdx.x * 256 + threadIdx.x) * 4;
  if (i + 4 <= n) {
    float4 f = *reinterpret_cast<const float4*>(in + i);
    ushort4 o;
    o.x = f2bf(f.x); o.y = f2bf(f.y); o.z = f2bf(f.z); o.w = f2bf(f.w);
    *reinterpret_cast<ushort4*>(out + i) = o;
  }
}

// ---------------------------------------------------------------------------
// C = act(A @ W^T + bias).  A: MxK bf16 row-major, W: NxK bf16 row-major.
// 256 threads (8 waves) -> 128x128 tile; wave = 32x64 (2x4 WMMA tiles).
// K-step 64. Tiles fetched by the Tensor Data Mover (wave 0 issues), double
// buffered so the TDM DMA overlaps the WMMA work of the previous step.
// ---------------------------------------------------------------------------
template <bool OUT_BF16, bool BIAS, bool SIG>
__global__ void __launch_bounds__(256)
gemm_xwt(const unsigned short* __restrict__ A, const unsigned short* __restrict__ Wt,
         const float* __restrict__ bias, void* __restrict__ out,
         int M, int Nn, int K) {
  __shared__ __align__(16) unsigned short As[2][128][72];  // 64 K + 8 pad
  __shared__ __align__(16) unsigned short Bs[2][128][72];

  const int tid  = threadIdx.x;
  const int lane = tid & 31;
  const int wave = tid >> 5;
  const int c16  = lane & 15;
  const int seg  = lane >> 4;
  const int wm   = wave >> 1;   // 0..3
  const int wn   = wave & 1;    // 0..1
  const int m0   = blockIdx.y * 128;
  const int n0   = blockIdx.x * 128;

  const v8f vzero = {};
  v8f acc[2][4];
  for (int mi = 0; mi < 2; ++mi)
    for (int ni = 0; ni < 4; ++ni) acc[mi][ni] = vzero;

  const unsigned short* gA0 = A  + (size_t)m0 * K;
  const unsigned short* gB0 = Wt + (size_t)n0 * K;
  const int nsteps = K / 64;

  if (wave == 0) {
    tdm_load_2d((unsigned)(uintptr_t)&As[0][0][0], gA0, K, M,  64, 128, K);
    tdm_load_2d((unsigned)(uintptr_t)&Bs[0][0][0], gB0, K, Nn, 64, 128, K);
  }

  for (int i = 0; i < nsteps; ++i) {
    if (wave == 0) wait_tensor0();
    __syncthreads();
    const int cur = i & 1;
    if (wave == 0 && i + 1 < nsteps) {
      const int nxt = cur ^ 1;
      tdm_load_2d((unsigned)(uintptr_t)&As[nxt][0][0], gA0 + (i + 1) * 64,
                  K, M, 64, 128, K);
      tdm_load_2d((unsigned)(uintptr_t)&Bs[nxt][0][0], gB0 + (i + 1) * 64,
                  K, Nn, 64, 128, K);
    }

    v16bf av[2][2], bv[4][2];
#pragma unroll
    for (int mi = 0; mi < 2; ++mi)
#pragma unroll
      for (int kh = 0; kh < 2; ++kh)
        av[mi][kh] = load_frag(&As[cur][wm * 32 + mi * 16 + c16][kh * 32 + seg * 16]);
#pragma unroll
    for (int ni = 0; ni < 4; ++ni)
#pragma unroll
      for (int kh = 0; kh < 2; ++kh)
        bv[ni][kh] = load_frag(&Bs[cur][wn * 64 + ni * 16 + c16][kh * 32 + seg * 16]);
#pragma unroll
    for (int mi = 0; mi < 2; ++mi)
#pragma unroll
      for (int ni = 0; ni < 4; ++ni) {
        acc[mi][ni] = wmma_bf16(av[mi][0], bv[ni][0], acc[mi][ni]);
        acc[mi][ni] = wmma_bf16(av[mi][1], bv[ni][1], acc[mi][ni]);
      }
  }

#pragma unroll
  for (int ni = 0; ni < 4; ++ni) {
    const int cn = n0 + wn * 64 + ni * 16 + c16;
    const float bias_v = BIAS ? bias[cn] : 0.0f;
#pragma unroll
    for (int mi = 0; mi < 2; ++mi) {
#pragma unroll
      for (int j = 0; j < 8; ++j) {
        const int row = m0 + wm * 32 + mi * 16 + j + 8 * seg;
        float val = acc[mi][ni][j] + bias_v;
        if (SIG) val = 1.0f / (1.0f + __expf(-val));
        if (OUT_BF16)
          reinterpret_cast<unsigned short*>(out)[(size_t)row * Nn + cn] = f2bf(val);
        else
          reinterpret_cast<float*>(out)[(size_t)row * Nn + cn] = val;
      }
    }
  }
}

// ---------------------------------------------------------------------------
// Fused flash attention per (head, 64-row block). 128 threads = 4 waves,
// each wave owns 16 query rows. N swept in 64-column K/V tiles, double
// buffered: K via async global->LDS copies, V prefetched into registers and
// transposed into LDS. Online softmax in exp2 domain with DPP reductions.
// Epilogue: o/rowsum * sigmoid-gate g, written as bf16.
// ---------------------------------------------------------------------------
__global__ void __launch_bounds__(128)
attn_flash(const unsigned short* __restrict__ q, const unsigned short* __restrict__ k,
           const unsigned short* __restrict__ v, const int* __restrict__ mask,
           const float* __restrict__ g, unsigned short* __restrict__ og) {
  __shared__ __align__(16) unsigned short Ks[2][64][72];     // (n, d) tiles
  __shared__ __align__(16) unsigned short Vt[2][64][72];     // transposed (d, n)
  __shared__ __align__(16) unsigned short Ps[4][16][72];     // per-wave P staging

  const int h    = blockIdx.y;
  const int tid  = threadIdx.x;
  const int lane = tid & 31;
  const int wave = tid >> 5;
  const int c16  = lane & 15;
  const int seg  = lane >> 4;
  const int rowbase = blockIdx.x * 64 + wave * 16;
  const int hc   = h * DHD;

  v16bf qa[2];
#pragma unroll
  for (int kh = 0; kh < 2; ++kh)
    qa[kh] = load_frag(q + (size_t)(rowbase + c16) * CH + hc + kh * 32 + seg * 16);

  float rmax[8], rsum[8];
  const v8f vzero = {};
  v8f oacc[4];
#pragma unroll
  for (int j = 0; j < 8; ++j) { rmax[j] = -1e30f; rsum[j] = 0.0f; }
#pragma unroll
  for (int t = 0; t < 4; ++t) oacc[t] = vzero;

  // exp2-domain softmax: fold log2(e) into scale and mask constants.
  const float scale2 = 0.125f * 1.44269504088896341f;
  const int lr = tid >> 1;          // 0..63: K/V row within tile
  const int lc = (tid & 1) * 32;    // 32-short d chunk
  const unsigned ldsK[2] = {(unsigned)(uintptr_t)&Ks[0][lr][lc],
                            (unsigned)(uintptr_t)&Ks[1][lr][lc]};
  const unsigned short* gK = k + (size_t)lr * CH + hc + lc;
  const unsigned short* gV = v + (size_t)lr * CH + hc + lc;
  const int nsteps = NK / 64;

  // prologue: start tile 0
  uint4 vreg[4];
#pragma unroll
  for (int c = 0; c < 4; ++c)
    vreg[c] = *reinterpret_cast<const uint4*>(gV + c * 8);
  async_copy64(ldsK[0], gK);

  for (int i = 0; i < nsteps; ++i) {
    const int n0  = i * 64;
    const int cur = i & 1;
    {  // stage this tile's V (prefetched last iteration) into LDS, transposed
      const unsigned short* e = reinterpret_cast<const unsigned short*>(vreg);
#pragma unroll
      for (int j = 0; j < 32; ++j) Vt[cur][lc + j][lr] = e[j];
    }
    wait_async0();
    __syncthreads();
    if (i + 1 < nsteps) {  // start next tile: async K, register-prefetch V
      async_copy64(ldsK[cur ^ 1], gK + (size_t)(n0 + 64) * CH);
#pragma unroll
      for (int c = 0; c < 4; ++c)
        vreg[c] = *reinterpret_cast<const uint4*>(gV + (size_t)(n0 + 64) * CH + c * 8);
    }

    float madd[4];
#pragma unroll
    for (int ti = 0; ti < 4; ++ti)
      madd[ti] = (float)(1 - mask[n0 + ti * 16 + c16]) * -1442695.0f;

    // scores: four 16x16 tiles over columns n0..n0+63, K = d = 64 (2 WMMA each)
    v8f sc[4];
#pragma unroll
    for (int ti = 0; ti < 4; ++ti) {
      v8f s = vzero;
      s = wmma_bf16(qa[0], load_frag(&Ks[cur][ti * 16 + c16][seg * 16]), s);
      s = wmma_bf16(qa[1], load_frag(&Ks[cur][ti * 16 + c16][32 + seg * 16]), s);
      sc[ti] = s;
    }

    float p[4][8];
#pragma unroll
    for (int j = 0; j < 8; ++j) {
      float x[4];
#pragma unroll
      for (int ti = 0; ti < 4; ++ti) x[ti] = sc[ti][j] * scale2 + madd[ti];
      float mx = fmaxf(fmaxf(x[0], x[1]), fmaxf(x[2], x[3]));
      mx = rowmax16(mx);
      const float nm   = fmaxf(rmax[j], mx);
      const float corr = fexp2(rmax[j] - nm);
      rmax[j] = nm;
      float ps = 0.0f;
#pragma unroll
      for (int ti = 0; ti < 4; ++ti) {
        x[ti] = fexp2(x[ti] - nm);
        p[ti][j] = x[ti];
        ps += x[ti];
      }
      ps = rowsum16(ps);
      rsum[j] = rsum[j] * corr + ps;
#pragma unroll
      for (int t = 0; t < 4; ++t) oacc[t][j] *= corr;
    }

    // C/D-layout P -> A-layout via per-wave LDS staging (same-wave DS in-order)
#pragma unroll
    for (int j = 0; j < 8; ++j)
#pragma unroll
      for (int ti = 0; ti < 4; ++ti)
        Ps[wave][j + 8 * seg][ti * 16 + c16] = f2bf(p[ti][j]);

    const v16bf pa0 = load_frag(&Ps[wave][c16][seg * 16]);        // K = n 0..31
    const v16bf pa1 = load_frag(&Ps[wave][c16][32 + seg * 16]);   // K = n 32..63
#pragma unroll
    for (int t = 0; t < 4; ++t) {
      v16bf vb0 = load_frag(&Vt[cur][t * 16 + c16][seg * 16]);
      v16bf vb1 = load_frag(&Vt[cur][t * 16 + c16][32 + seg * 16]);
      oacc[t] = wmma_bf16(pa0, vb0, oacc[t]);
      oacc[t] = wmma_bf16(pa1, vb1, oacc[t]);
    }
  }

  // epilogue: normalize, gate, store bf16
#pragma unroll
  for (int t = 0; t < 4; ++t) {
    const int cc = hc + t * 16 + c16;
#pragma unroll
    for (int j = 0; j < 8; ++j) {
      const int row = rowbase + j + 8 * seg;
      float val = oacc[t][j] / rsum[j];
      val *= g[(size_t)row * CH + cc];
      og[(size_t)row * CH + cc] = f2bf(val);
    }
  }
}

// ---------------------------------------------------------------------------
extern "C" void kernel_launch(void* const* d_in, const int* in_sizes, int n_in,
                              void* d_out, int out_size, void* d_ws, size_t ws_size,
                              hipStream_t stream) {
  const float* s    = (const float*)d_in[0];
  const int*   mask = (const int*)d_in[1];
  const float* k_in = (const float*)d_in[2];
  const float* Wq   = (const float*)d_in[3];
  const float* bq   = (const float*)d_in[4];
  const float* Wk   = (const float*)d_in[5];
  const float* Wv   = (const float*)d_in[6];
  const float* Wg   = (const float*)d_in[7];
  const float* Wo   = (const float*)d_in[8];
  float* out = (float*)d_out;

  const size_t SC = (size_t)SQ * CH;  // 2M elems
  const size_t CC = (size_t)CH * CH;  // 1M elems

  char* ws = (char*)d_ws;
  unsigned short* s_bf   = (unsigned short*)ws;            ws += SC * 2;
  unsigned short* kin_bf = (unsigned short*)ws;            ws += SC * 2;
  unsigned short* wq_bf  = (unsigned short*)ws;            ws += CC * 2;
  unsigned short* wk_bf  = (unsigned short*)ws;            ws += CC * 2;
  unsigned short* wv_bf  = (unsigned short*)ws;            ws += CC * 2;
  unsigned short* wg_bf  = (unsigned short*)ws;            ws += CC * 2;
  unsigned short* wo_bf  = (unsigned short*)ws;            ws += CC * 2;
  unsigned short* q_bf   = (unsigned short*)ws;            ws += SC * 2;
  unsigned short* k_bf   = (unsigned short*)ws;            ws += SC * 2;
  unsigned short* v_bf   = (unsigned short*)ws;            ws += SC * 2;
  float*          g_f32  = (float*)ws;                     ws += SC * 4;
  unsigned short* og_bf  = (unsigned short*)ws;            ws += SC * 2;

  // fp32 -> bf16 staging
  cvt_f32_bf16<<<dim3((SC / 4 + 255) / 256), 256, 0, stream>>>(s, s_bf, (int)SC);
  cvt_f32_bf16<<<dim3((SC / 4 + 255) / 256), 256, 0, stream>>>(k_in, kin_bf, (int)SC);
  cvt_f32_bf16<<<dim3((CC / 4 + 255) / 256), 256, 0, stream>>>(Wq, wq_bf, (int)CC);
  cvt_f32_bf16<<<dim3((CC / 4 + 255) / 256), 256, 0, stream>>>(Wk, wk_bf, (int)CC);
  cvt_f32_bf16<<<dim3((CC / 4 + 255) / 256), 256, 0, stream>>>(Wv, wv_bf, (int)CC);
  cvt_f32_bf16<<<dim3((CC / 4 + 255) / 256), 256, 0, stream>>>(Wg, wg_bf, (int)CC);
  cvt_f32_bf16<<<dim3((CC / 4 + 255) / 256), 256, 0, stream>>>(Wo, wo_bf, (int)CC);

  const dim3 ggrid(CH / 128, SQ / 128);
  const dim3 gblk(256);

  // projections
  gemm_xwt<true, true,  false><<<ggrid, gblk, 0, stream>>>(s_bf,   wq_bf, bq,      q_bf,  SQ, CH, CH);
  gemm_xwt<true, false, false><<<ggrid, gblk, 0, stream>>>(kin_bf, wk_bf, nullptr, k_bf,  NK, CH, CH);
  gemm_xwt<true, false, false><<<ggrid, gblk, 0, stream>>>(kin_bf, wv_bf, nullptr, v_bf,  NK, CH, CH);
  gemm_xwt<false, false, true><<<ggrid, gblk, 0, stream>>>(s_bf,   wg_bf, nullptr, g_f32, SQ, CH, CH);

  // fused masked-softmax attention + gating
  attn_flash<<<dim3(SQ / 64, NH), 128, 0, stream>>>(q_bf, k_bf, v_bf, mask, g_f32, og_bf);

  // output projection (fp32 result)
  gemm_xwt<false, false, false><<<ggrid, gblk, 0, stream>>>(og_bf, wo_bf, nullptr, out, SQ, CH, CH);
}
